// GraphTransformer_25829933318820
// MI455X (gfx1250) — matrix operand order repaired and checked by reading, not compiled, not run on hardware
//
#include <hip/hip_runtime.h>
#include <hip/hip_bf16.h>
#include <math.h>

// ---------------------------------------------------------------------------
// Problem constants (from reference setup_inputs)
// ---------------------------------------------------------------------------
#define NN   32768      // nodes
#define MM   131072     // subgraph nodes
#define EE   524288     // edges
#define EO   262144     // original edges
#define BB   16         // batch
#define PP   256        // patches
#define BP   4096       // B*P
#define SS   257        // sequence (P+1)
#define RR   4112       // S*B rows
#define DD   128        // model dim
#define FFD  256        // feedforward dim
#define HIDD 256
#define OUTD 64

typedef _Float16 v16h __attribute__((ext_vector_type(16)));
typedef _Float16 v8h  __attribute__((ext_vector_type(8)));
typedef float    v8f  __attribute__((ext_vector_type(8)));
typedef float    v4f  __attribute__((ext_vector_type(4)));
typedef int      aint4 __attribute__((vector_size(16)));   // matches builtin param type

#if __has_builtin(__builtin_amdgcn_global_load_async_to_lds_b128) && \
    __has_builtin(__builtin_amdgcn_s_wait_asynccnt)
#define HAVE_ASYNC_LDS 1
typedef __attribute__((address_space(1))) aint4* g_aint4_p;   // global int4*
typedef __attribute__((address_space(3))) aint4* l_aint4_p;   // LDS int4*
#else
#define HAVE_ASYNC_LDS 0
#endif

// ---------------------------------------------------------------------------
// Weight packing: fp32 [K,N] (or transposed [N,K]) -> f16 in contiguous-k WMMA-B
// per-lane layout.  Fragment (kt,nt) occupies 32 lanes x 16 halves contiguous:
//   lane L, half e:  K = kt*32 + (L>>4)*16 + e,  Nc = nt*16 + (L&15)
// ---------------------------------------------------------------------------
__global__ void pack_weight_kernel(const float* __restrict__ W, _Float16* __restrict__ out,
                                   int Kin, int N, int transpose) {
    int idx = blockIdx.x * 256 + threadIdx.x;
    int ntiles = N >> 4;
    int ktiles = (Kin + 31) >> 5;
    int total  = ktiles * ntiles * 512;
    if (idx >= total) return;
    int e    = idx & 15;
    int lane = (idx >> 4) & 31;
    int tile = idx >> 9;
    int nt = tile % ntiles;
    int kt = tile / ntiles;
    int k  = kt * 32 + (lane >> 4) * 16 + e;
    int n  = nt * 16 + (lane & 15);
    float v = 0.f;
    if (k < Kin) v = transpose ? W[(size_t)n * Kin + k] : W[(size_t)k * N + n];
    out[idx] = (_Float16)v;
}

// ---------------------------------------------------------------------------
// Generic WMMA GEMM:  C[M,N] = act(A[M,Kin] @ B + bias) (+ residual)
// Requires Kin % 32 == 0 (callers pad feature buffers).
// block = 256 threads = 8 waves; tile = 64(M) x 64(N);
// each wave owns two 16x16 M-tiles sharing one B fragment -> 2 WMMA / k-step.
// A slab staged to LDS via CDNA5 async global->LDS DMA (ASYNCcnt).
// act: 0 = none, 1 = relu, 2 = exact gelu
// ---------------------------------------------------------------------------
__global__ void __launch_bounds__(256) gemm_wmma_kernel(
    const float* __restrict__ A, int lda,
    const _Float16* __restrict__ Bp,
    const float* __restrict__ bias,
    const float* residual,
    float* C,
    int M, int Kin, int N, int act)
{
    __shared__ float As[64][36];      // row stride 144B (16B aligned)
    const int tid  = threadIdx.x;
    const int lane = tid & 31;
    const int wid  = tid >> 5;
    const int mq   = wid >> 2;        // 0..1 (pair of M-tiles)
    const int nw   = wid & 3;         // 0..3 (N-tile)
    const int ntiles_n = N >> 4;
    const int m0 = blockIdx.y * 64;
    const int n0 = blockIdx.x * 64;
    const int ksteps = Kin >> 5;
    const int nt = (n0 >> 4) + nw;

    // staging coords: 64 rows x 32 cols fp32, 8 floats / thread
    const int srow = tid >> 2;
    const int scol = (tid & 3) * 8;
    int gm_s = m0 + srow; if (gm_s >= M) gm_s = M - 1;
    const float* aptr = A + (size_t)gm_s * lda + scol;

    // fragment coords (contiguous k order: K = (lane>>4)*16 + e, matches packed B)
    const int mrow0 = mq * 32 + (lane & 15);
    const int mrow1 = mrow0 + 16;
    const int ksel  = (lane >> 4) * 16;

    v8f acc0 = {}, acc1 = {};

    for (int ks = 0; ks < ksteps; ++ks) {
        const int k0 = ks << 5;
#if HAVE_ASYNC_LDS
        {
            // CDNA5 async global->LDS DMA (tracked by ASYNCcnt)
            __builtin_amdgcn_global_load_async_to_lds_b128(
                (g_aint4_p)(aptr + k0), (l_aint4_p)&As[srow][scol], 0, 0);
            __builtin_amdgcn_global_load_async_to_lds_b128(
                (g_aint4_p)(aptr + k0 + 4), (l_aint4_p)&As[srow][scol + 4], 0, 0);
            __builtin_amdgcn_s_wait_asynccnt(0);
        }
#else
        {
            v4f x0 = *(const v4f*)(aptr + k0);
            v4f x1 = *(const v4f*)(aptr + k0 + 4);
            *(v4f*)&As[srow][scol]     = x0;
            *(v4f*)&As[srow][scol + 4] = x1;
        }
#endif
        __syncthreads();

        v8f a00 = *(const v8f*)&As[mrow0][ksel];
        v8f a01 = *(const v8f*)&As[mrow0][ksel + 8];
        v8f a10 = *(const v8f*)&As[mrow1][ksel];
        v8f a11 = *(const v8f*)&As[mrow1][ksel + 8];
        v16h af0, af1;
        #pragma unroll
        for (int i = 0; i < 8; ++i) {
            af0[i]     = (_Float16)a00[i];
            af0[8 + i] = (_Float16)a01[i];
            af1[i]     = (_Float16)a10[i];
            af1[8 + i] = (_Float16)a11[i];
        }
        const _Float16* bptr = Bp + (((size_t)ks * ntiles_n + nt) * 32 + lane) * 16;
        v16h bf = *(const v16h*)bptr;
        if (ks + 1 < ksteps)
            __builtin_prefetch(Bp + (((size_t)(ks + 1) * ntiles_n + nt) * 32 + lane) * 16, 0, 1);

        acc0 = __builtin_amdgcn_wmma_f32_16x16x32_f16(false, af0, false, bf, (short)0, acc0, false, false);
        acc1 = __builtin_amdgcn_wmma_f32_16x16x32_f16(false, af1, false, bf, (short)0, acc1, false, false);
        __syncthreads();
    }

    // ---- epilogue: C layout (VGPR r: M = base + r, lanes 16-31 -> M+8) ----
    const int gn = n0 + nw * 16 + (lane & 15);
    const float bv = bias ? bias[gn] : 0.0f;
    #pragma unroll
    for (int half = 0; half < 2; ++half) {
        v8f acc = half ? acc1 : acc0;
        const int mbase = m0 + (mq * 2 + half) * 16 + (lane >> 4) * 8;
        #pragma unroll
        for (int r = 0; r < 8; ++r) {
            int gm = mbase + r;
            if (gm < M) {
                float v = acc[r] + bv;
                if (act == 1)      v = fmaxf(v, 0.0f);
                else if (act == 2) v = 0.5f * v * (1.0f + erff(v * 0.70710678118654752f));
                if (residual)      v += residual[(size_t)gm * N + gn];
                C[(size_t)gm * N + gn] = v;
            }
        }
    }
}

// ---------------------------------------------------------------------------
// Feature builders (NeRF positional encoding), K padded to multiples of 32
// ---------------------------------------------------------------------------
__global__ void nodefeat_kernel(const float* __restrict__ x, const float* __restrict__ pos,
                                float* __restrict__ out) {
    size_t gid = (size_t)blockIdx.x * 256 + threadIdx.x;
    if (gid >= (size_t)NN * 64) return;
    int c = (int)(gid & 63);
    size_t n = gid >> 6;
    float v = 0.0f;
    if (c < 16) {
        v = x[n * 16 + c];
    } else if (c < 40) {
        int cc = c - 16;
        int coord = cc >> 3, r = cc & 7, i = r & 3, isc = r >> 2;
        float ang = pos[n * 3 + coord] * ((float)(1 << i) * 3.14159265358979323846f);
        v = isc ? cosf(ang) : sinf(ang);
    }
    out[gid] = v;
}

__global__ void patchfeat_kernel(const float* __restrict__ ppos, float* __restrict__ out) {
    size_t gid = (size_t)blockIdx.x * 256 + threadIdx.x;
    if (gid >= (size_t)BP * 32) return;
    int c = (int)(gid & 31);
    size_t n = gid >> 5;
    float v = 0.0f;
    if (c < 24) {
        int coord = c >> 3, r = c & 7, i = r & 3, isc = r >> 2;
        float ang = ppos[n * 3 + coord] * ((float)(1 << i) * 3.14159265358979323846f);
        v = isc ? cosf(ang) : sinf(ang);
    }
    out[gid] = v;
}

// ---------------------------------------------------------------------------
// Graph scatter / gather primitives (D = 128, float4 per thread)
// ---------------------------------------------------------------------------
__global__ void gather_rows_kernel(const float* __restrict__ src, const int* __restrict__ idx,
                                   float* __restrict__ out, size_t rows) {
    size_t gid = (size_t)blockIdx.x * 256 + threadIdx.x;
    size_t j = gid >> 5; if (j >= rows) return;
    int dq = (int)(gid & 31) * 4;
    const float* s = src + (size_t)idx[j] * DD + dq;
    float* d = out + j * DD + dq;
    d[0] = s[0]; d[1] = s[1]; d[2] = s[2]; d[3] = s[3];
}

__global__ void scatter_add_kernel(const float* __restrict__ src, const int* __restrict__ idx,
                                   float* sums, float* cnt, size_t rows) {
    size_t gid = (size_t)blockIdx.x * 256 + threadIdx.x;
    size_t j = gid >> 5; if (j >= rows) return;
    int dq = (int)(gid & 31) * 4;
    int p = idx[j];
    const float* s = src + j * DD + dq;
    float* d = sums + (size_t)p * DD + dq;
    atomicAdd(d + 0, s[0]); atomicAdd(d + 1, s[1]);
    atomicAdd(d + 2, s[2]); atomicAdd(d + 3, s[3]);
    if (dq == 0) atomicAdd(cnt + p, 1.0f);
}

__global__ void mean_kernel(const float* __restrict__ sums, const float* __restrict__ cnt,
                            float* __restrict__ out, size_t rows) {
    size_t gid = (size_t)blockIdx.x * 256 + threadIdx.x;
    size_t r = gid >> 7; if (r >= rows) return;
    int c = (int)(gid & 127);
    out[r * DD + c] = sums[r * DD + c] / fmaxf(cnt[r], 1.0f);
}

__global__ void gather_mean_kernel(const float* __restrict__ sums, const float* __restrict__ cnt,
                                   const int* __restrict__ idx, float* __restrict__ out, size_t rows) {
    size_t gid = (size_t)blockIdx.x * 256 + threadIdx.x;
    size_t j = gid >> 5; if (j >= rows) return;
    int dq = (int)(gid & 31) * 4;
    int p = idx[j];
    float inv = 1.0f / fmaxf(cnt[p], 1.0f);
    const float* s = sums + (size_t)p * DD + dq;
    float* d = out + j * DD + dq;
    d[0] = s[0] * inv; d[1] = s[1] * inv; d[2] = s[2] * inv; d[3] = s[3] * inv;
}

__global__ void degree_kernel(const int* __restrict__ src, const int* __restrict__ dst,
                              float* degs, float* degd) {
    size_t j = (size_t)blockIdx.x * 256 + threadIdx.x;
    if (j >= EE) return;
    atomicAdd(degs + src[j], 1.0f);
    atomicAdd(degd + dst[j], 1.0f);
}

// agg[dst] += (xm[src] + edge_enc(edge_attr[emap])) * enorm ; edge encoder fused.
__global__ void message_kernel(const float* __restrict__ xm,
                               const int* __restrict__ src, const int* __restrict__ dst,
                               const int* __restrict__ emap,
                               const float* __restrict__ eattr,
                               const float* __restrict__ ew, const float* __restrict__ eb,
                               const float* __restrict__ degs, const float* __restrict__ degd,
                               float* agg) {
    size_t gid = (size_t)blockIdx.x * 256 + threadIdx.x;
    size_t j = gid >> 5; if (j >= (size_t)EE) return;
    int dq = (int)(gid & 31) * 4;
    int sj = src[j], dj = dst[j], ej = emap[j];
    float norm = rsqrtf(fmaxf(degs[sj], 1.0f) * fmaxf(degd[dj], 1.0f));
    v4f ev = { eb[dq], eb[dq + 1], eb[dq + 2], eb[dq + 3] };
    const float* ar = eattr + (size_t)ej * 16;
    #pragma unroll
    for (int k = 0; k < 16; ++k) {
        float a = ar[k];
        const float* wr = ew + k * DD + dq;
        ev[0] = fmaf(a, wr[0], ev[0]); ev[1] = fmaf(a, wr[1], ev[1]);
        ev[2] = fmaf(a, wr[2], ev[2]); ev[3] = fmaf(a, wr[3], ev[3]);
    }
    const float* xs = xm + (size_t)sj * DD + dq;
    float* d = agg + (size_t)dj * DD + dq;
    atomicAdd(d + 0, (xs[0] + ev[0]) * norm);
    atomicAdd(d + 1, (xs[1] + ev[1]) * norm);
    atomicAdd(d + 2, (xs[2] + ev[2]) * norm);
    atomicAdd(d + 3, (xs[3] + ev[3]) * norm);
}

// ---------------------------------------------------------------------------
// BatchNorm over M rows (column stats via atomics) then xm += relu(bn(u))
// ---------------------------------------------------------------------------
__global__ void colstats_kernel(const float* __restrict__ X, float* sum, float* sumsq, int rows) {
    int col = threadIdx.x & 127;
    int rh  = threadIdx.x >> 7;
    int r0  = blockIdx.x * 256 + rh * 128;
    int r1  = r0 + 128; if (r1 > rows) r1 = rows;
    float s = 0.f, q = 0.f;
    for (int r = r0; r < r1; ++r) {
        float v = X[(size_t)r * DD + col];
        s += v; q += v * v;
    }
    atomicAdd(sum + col, s);
    atomicAdd(sumsq + col, q);
}

__global__ void bn_relu_add_kernel(const float* __restrict__ u,
                                   const float* __restrict__ sum, const float* __restrict__ sumsq,
                                   const float* __restrict__ g, const float* __restrict__ b,
                                   float* xm, int rows) {
    size_t gid = (size_t)blockIdx.x * 256 + threadIdx.x;
    size_t r = gid >> 7; if (r >= (size_t)rows) return;
    int c = (int)(gid & 127);
    float mu  = sum[c] / (float)rows;
    float var = sumsq[c] / (float)rows - mu * mu;
    float v = (u[r * DD + c] - mu) * rsqrtf(var + 1e-5f) * g[c] + b[c];
    xm[r * DD + c] += fmaxf(v, 0.0f);
}

// ---------------------------------------------------------------------------
// Transformer helpers
// ---------------------------------------------------------------------------
__global__ void tbuild_kernel(const float* __restrict__ sub, const float* __restrict__ cls,
                              float* __restrict__ t) {
    size_t gid = (size_t)blockIdx.x * 256 + threadIdx.x;
    if (gid >= (size_t)RR * DD) return;
    int c = (int)(gid & 127);
    size_t row = gid >> 7;
    int s = (int)(row / BB), b = (int)(row % BB);
    t[gid] = (s == 0) ? cls[c] : sub[((size_t)b * PP + (s - 1)) * DD + c];
}

// LayerNorm, one wave32 per row (D = 128 -> 4 elements/lane)
__global__ void __launch_bounds__(256) ln_kernel(const float* __restrict__ X,
                                                 const float* __restrict__ g,
                                                 const float* __restrict__ b,
                                                 float* __restrict__ Y, int rows) {
    int row  = blockIdx.x * 8 + (threadIdx.x >> 5);
    int lane = threadIdx.x & 31;
    if (row >= rows) return;
    const float* x = X + (size_t)row * DD;
    v4f v = *(const v4f*)(x + lane * 4);
    float s = v[0] + v[1] + v[2] + v[3];
    for (int off = 16; off; off >>= 1) s += __shfl_xor(s, off);
    float mu = s * (1.0f / 128.0f);
    float q = 0.f;
    #pragma unroll
    for (int i = 0; i < 4; ++i) { float d = v[i] - mu; q += d * d; }
    for (int off = 16; off; off >>= 1) q += __shfl_xor(q, off);
    float rstd = rsqrtf(q * (1.0f / 128.0f) + 1e-5f);
    float* y = Y + (size_t)row * DD;
    #pragma unroll
    for (int i = 0; i < 4; ++i) {
        int c = lane * 4 + i;
        y[c] = (v[i] - mu) * rstd * g[c] + b[c];
    }
}

// One block per (b*h, s): scores over 257 keys, softmax, weighted sum of V.
__global__ void __launch_bounds__(256) attn_kernel(const float* __restrict__ qkv,
                                                   float* __restrict__ o) {
    const int bh = blockIdx.x;
    const int b = bh >> 2, h = bh & 3;
    const int s = blockIdx.y;
    const int tid = threadIdx.x;
    __shared__ float sc[SS];
    __shared__ float qs[32];
    __shared__ float red[9];
    if (tid < 32) qs[tid] = qkv[((size_t)s * BB + b) * 384 + h * 32 + tid];
    __syncthreads();
    for (int t = tid; t < SS; t += 256) {
        const float* kr = qkv + ((size_t)t * BB + b) * 384 + 128 + h * 32;
        float d = 0.f;
        #pragma unroll
        for (int i = 0; i < 32; ++i) d = fmaf(qs[i], kr[i], d);
        d *= 0.17677669529663687f;            // 1/sqrt(32)
        if (t > s && s != 0) d = -1e9f;       // causal mask, row 0 sees all
        sc[t] = d;
    }
    __syncthreads();
    float m = -3.4e38f;
    for (int t = tid; t < SS; t += 256) m = fmaxf(m, sc[t]);
    for (int off = 16; off; off >>= 1) m = fmaxf(m, __shfl_xor(m, off));
    if ((tid & 31) == 0) red[tid >> 5] = m;
    __syncthreads();
    if (tid == 0) {
        float mm = red[0];
        for (int w = 1; w < 8; ++w) mm = fmaxf(mm, red[w]);
        red[8] = mm;
    }
    __syncthreads();
    m = red[8];
    float ps = 0.f;
    for (int t = tid; t < SS; t += 256) { float e = __expf(sc[t] - m); sc[t] = e; ps += e; }
    for (int off = 16; off; off >>= 1) ps += __shfl_xor(ps, off);
    __syncthreads();
    if ((tid & 31) == 0) red[tid >> 5] = ps;
    __syncthreads();
    if (tid == 0) {
        float ss = 0.f;
        for (int w = 0; w < 8; ++w) ss += red[w];
        red[8] = ss;
    }
    __syncthreads();
    float inv = 1.0f / red[8];
    if (tid < 32) {
        float acc = 0.f;
        for (int t = 0; t < SS; ++t)
            acc = fmaf(sc[t], qkv[((size_t)t * BB + b) * 384 + 256 + h * 32 + tid], acc);
        o[((size_t)s * BB + b) * DD + h * 32 + tid] = acc * inv;
    }
}

// Head BN over 16 rows, fused exact GELU, in place.
__global__ void bnhead_kernel(float* X, const float* __restrict__ g,
                              const float* __restrict__ b, int C) {
    int c = blockIdx.x * 256 + threadIdx.x;
    if (c >= C) return;
    float s = 0.f, q = 0.f;
    for (int r = 0; r < 16; ++r) { float v = X[(size_t)r * C + c]; s += v; q += v * v; }
    float mu = s * (1.0f / 16.0f);
    float var = q * (1.0f / 16.0f) - mu * mu;
    float rstd = rsqrtf(var + 1e-5f);
    for (int r = 0; r < 16; ++r) {
        float v = (X[(size_t)r * C + c] - mu) * rstd * g[c] + b[c];
        X[(size_t)r * C + c] = 0.5f * v * (1.0f + erff(v * 0.70710678118654752f));
    }
}

// ---------------------------------------------------------------------------
// Host orchestration
// ---------------------------------------------------------------------------
enum {
    IN_X = 0, IN_EDGE_ATTR, IN_POS, IN_PATCH_POS, IN_CSG, IN_SBATCH, IN_NMAP, IN_EMAP, IN_MASK,
    P_NODE_W, P_NODE_B, P_EDGE_W, P_EDGE_B, P_GNN_W, P_GNN_B, P_U_W, P_U_B, P_U_G, P_U_BB,
    P_SPE_W, P_SPE_B, P_CLS, P_T_INW, P_T_INB, P_T_OW, P_T_OB, P_T_W1, P_T_B1, P_T_W2, P_T_B2,
    P_N1G, P_N1B, P_N2G, P_N2B, P_D1W, P_D1B, P_BN1G, P_BN1B, P_D2W, P_D2B, P_BN2G, P_BN2B,
    P_D3W, P_D3B
};

extern "C" void kernel_launch(void* const* d_in, const int* in_sizes, int n_in,
                              void* d_out, int out_size, void* d_ws, size_t ws_size,
                              hipStream_t stream) {
    (void)in_sizes; (void)n_in; (void)out_size; (void)ws_size;

    const float* x      = (const float*)d_in[IN_X];
    const float* eattr  = (const float*)d_in[IN_EDGE_ATTR];
    const float* pos    = (const float*)d_in[IN_POS];
    const float* ppos   = (const float*)d_in[IN_PATCH_POS];
    const int*   csg    = (const int*)d_in[IN_CSG];
    const int*   src    = csg;
    const int*   dst    = csg + EE;
    const int*   sbatch = (const int*)d_in[IN_SBATCH];
    const int*   nmap   = (const int*)d_in[IN_NMAP];
    const int*   emap   = (const int*)d_in[IN_EMAP];

    const float* node_w = (const float*)d_in[P_NODE_W];
    const float* node_b = (const float*)d_in[P_NODE_B];
    const float* edge_w = (const float*)d_in[P_EDGE_W];
    const float* edge_b = (const float*)d_in[P_EDGE_B];
    const float* gnn_w  = (const float*)d_in[P_GNN_W];
    const float* gnn_b  = (const float*)d_in[P_GNN_B];
    const float* U_w    = (const float*)d_in[P_U_W];
    const float* U_b    = (const float*)d_in[P_U_B];
    const float* U_g    = (const float*)d_in[P_U_G];
    const float* U_bb   = (const float*)d_in[P_U_BB];
    const float* spe_w  = (const float*)d_in[P_SPE_W];
    const float* spe_b  = (const float*)d_in[P_SPE_B];
    const float* cls    = (const float*)d_in[P_CLS];
    const float* t_inw  = (const float*)d_in[P_T_INW];
    const float* t_inb  = (const float*)d_in[P_T_INB];
    const float* t_ow   = (const float*)d_in[P_T_OW];
    const float* t_ob   = (const float*)d_in[P_T_OB];
    const float* t_w1   = (const float*)d_in[P_T_W1];
    const float* t_b1   = (const float*)d_in[P_T_B1];
    const float* t_w2   = (const float*)d_in[P_T_W2];
    const float* t_b2   = (const float*)d_in[P_T_B2];
    const float* n1g    = (const float*)d_in[P_N1G];
    const float* n1b    = (const float*)d_in[P_N1B];
    const float* n2g    = (const float*)d_in[P_N2G];
    const float* n2b    = (const float*)d_in[P_N2B];
    const float* d1w    = (const float*)d_in[P_D1W];
    const float* d1b    = (const float*)d_in[P_D1B];
    const float* bn1g   = (const float*)d_in[P_BN1G];
    const float* bn1b   = (const float*)d_in[P_BN1B];
    const float* d2w    = (const float*)d_in[P_D2W];
    const float* d2b    = (const float*)d_in[P_D2B];
    const float* bn2g   = (const float*)d_in[P_BN2G];
    const float* bn2b   = (const float*)d_in[P_BN2B];
    const float* d3w    = (const float*)d_in[P_D3W];
    const float* d3b    = (const float*)d_in[P_D3B];

    // ---- workspace carve ----
    char* base = (char*)d_ws;
    size_t cur = 0;
    auto alloc = [&](size_t bytes) -> void* {
        void* p = base + cur;
        cur += (bytes + 255) & ~(size_t)255;
        return p;
    };
    float* f_nodefeat  = (float*)alloc((size_t)NN * 64 * 4);
    float* f_h         = (float*)alloc((size_t)NN * DD * 4);
    float* f_bufA      = (float*)alloc((size_t)MM * DD * 4);   // xm
    float* f_bufB      = (float*)alloc((size_t)MM * DD * 4);   // tmp / u
    float* f_agg       = (float*)alloc((size_t)MM * DD * 4);
    float* f_psum      = (float*)alloc((size_t)BP * DD * 4);
    float* f_pcnt      = (float*)alloc((size_t)BP * 4);
    float* f_nsum      = (float*)alloc((size_t)NN * DD * 4);
    float* f_ncnt      = (float*)alloc((size_t)NN * 4);
    float* f_degs      = (float*)alloc((size_t)MM * 4);
    float* f_degd      = (float*)alloc((size_t)MM * 4);
    float* f_patchfeat = (float*)alloc((size_t)BP * 32 * 4);
    float* f_sub       = (float*)alloc((size_t)BP * DD * 4);
    float* f_sub2      = (float*)alloc((size_t)BP * DD * 4);
    float* f_t         = (float*)alloc((size_t)RR * DD * 4);
    float* f_hln       = (float*)alloc((size_t)RR * DD * 4);
    float* f_qkv       = (float*)alloc((size_t)RR * 384 * 4);
    float* f_o         = (float*)alloc((size_t)RR * DD * 4);
    float* f_ff        = (float*)alloc((size_t)RR * FFD * 4);
    float* f_h1        = (float*)alloc((size_t)16 * HIDD * 4);
    float* f_h2        = (float*)alloc((size_t)16 * HIDD * 4);
    float* f_stats     = (float*)alloc(256 * 4);

    _Float16* p_node = (_Float16*)alloc(2 * 8 * 512 * 2);
    _Float16* p_gnn[3]; for (int i = 0; i < 3; ++i) p_gnn[i] = (_Float16*)alloc(4 * 8 * 512 * 2);
    _Float16* p_U[2];   for (int i = 0; i < 2; ++i) p_U[i]   = (_Float16*)alloc(4 * 8 * 512 * 2);
    _Float16* p_spe  = (_Float16*)alloc(1 * 8 * 512 * 2);
    _Float16* p_qkvw[4]; for (int i = 0; i < 4; ++i) p_qkvw[i] = (_Float16*)alloc(4 * 24 * 512 * 2);
    _Float16* p_ow[4];   for (int i = 0; i < 4; ++i) p_ow[i]   = (_Float16*)alloc(4 * 8 * 512 * 2);
    _Float16* p_w1[4];   for (int i = 0; i < 4; ++i) p_w1[i]   = (_Float16*)alloc(4 * 16 * 512 * 2);
    _Float16* p_w2[4];   for (int i = 0; i < 4; ++i) p_w2[i]   = (_Float16*)alloc(8 * 8 * 512 * 2);
    _Float16* p_d1 = (_Float16*)alloc(4 * 16 * 512 * 2);
    _Float16* p_d2 = (_Float16*)alloc(8 * 16 * 512 * 2);
    _Float16* p_d3 = (_Float16*)alloc(8 * 4 * 512 * 2);

    auto pack = [&](const float* W, _Float16* outp, int Kin, int Nc, int tr) {
        int total = ((Kin + 31) >> 5) * (Nc >> 4) * 512;
        pack_weight_kernel<<<dim3((total + 255) / 256), dim3(256), 0, stream>>>(W, outp, Kin, Nc, tr);
    };
    auto gemm = [&](const float* A, int lda, const _Float16* Bp, const float* bias,
                    const float* res, float* Cc, int Mr, int Kin, int Nc, int act) {
        dim3 g(Nc / 64, (Mr + 63) / 64);
        gemm_wmma_kernel<<<g, dim3(256), 0, stream>>>(A, lda, Bp, bias, res, Cc, Mr, Kin, Nc, act);
    };

    // ---- pack all weights (f16 WMMA-B layout, zero-padded K) ----
    pack(node_w, p_node, 40, DD, 0);
    for (int i = 0; i < 3; ++i) pack(gnn_w + (size_t)i * DD * DD, p_gnn[i], DD, DD, 0);
    for (int i = 0; i < 2; ++i) pack(U_w + (size_t)i * DD * DD, p_U[i], DD, DD, 0);
    pack(spe_w, p_spe, 24, DD, 0);
    for (int l = 0; l < 4; ++l) {
        pack(t_inw + (size_t)l * 384 * DD, p_qkvw[l], DD, 384, 1);   // (3D,D) -> B = W^T
        pack(t_ow + (size_t)l * DD * DD, p_ow[l], DD, DD, 1);
        pack(t_w1 + (size_t)l * DD * FFD, p_w1[l], DD, FFD, 0);
        pack(t_w2 + (size_t)l * FFD * DD, p_w2[l], FFD, DD, 0);
    }
    pack(d1w, p_d1, DD, HIDD, 0);
    pack(d2w, p_d2, HIDD, HIDD, 0);
    pack(d3w, p_d3, HIDD, OUTD, 0);

    // ---- encoders ----
    nodefeat_kernel<<<dim3(((size_t)NN * 64 + 255) / 256), dim3(256), 0, stream>>>(x, pos, f_nodefeat);
    gemm(f_nodefeat, 64, p_node, node_b, nullptr, f_h, NN, 64, DD, 0);

    // xm = h[nmap]
    gather_rows_kernel<<<dim3(((size_t)MM * 32 + 255) / 256), dim3(256), 0, stream>>>(f_h, nmap, f_bufA, MM);

    // degrees
    (void)hipMemsetAsync(f_degs, 0, (size_t)MM * 4, stream);
    (void)hipMemsetAsync(f_degd, 0, (size_t)MM * 4, stream);
    degree_kernel<<<dim3((EE + 255) / 256), dim3(256), 0, stream>>>(src, dst, f_degs, f_degd);

    float* xm  = f_bufA;
    float* tmp = f_bufB;
    const dim3 gM32(((size_t)MM * 32 + 255) / 256), b256(256);

    for (int i = 0; i < 3; ++i) {
        if (i > 0) {
            // sub = seg_mean(xm, batch); u = gather(sub @ U_w + U_b); xm += relu(bn(u))
            (void)hipMemsetAsync(f_psum, 0, (size_t)BP * DD * 4, stream);
            (void)hipMemsetAsync(f_pcnt, 0, (size_t)BP * 4, stream);
            scatter_add_kernel<<<gM32, b256, 0, stream>>>(xm, sbatch, f_psum, f_pcnt, MM);
            mean_kernel<<<dim3(((size_t)BP * DD + 255) / 256), b256, 0, stream>>>(f_psum, f_pcnt, f_sub, BP);
            gemm(f_sub, DD, p_U[i - 1], U_b + (size_t)(i - 1) * DD, nullptr, f_sub2, BP, DD, DD, 0);
            gather_rows_kernel<<<gM32, b256, 0, stream>>>(f_sub2, sbatch, tmp, MM);
            (void)hipMemsetAsync(f_stats, 0, 256 * 4, stream);
            colstats_kernel<<<dim3((MM + 255) / 256), b256, 0, stream>>>(tmp, f_stats, f_stats + 128, MM);
            bn_relu_add_kernel<<<dim3(((size_t)MM * DD + 255) / 256), b256, 0, stream>>>(
                tmp, f_stats, f_stats + 128, U_g + (size_t)(i - 1) * DD, U_bb + (size_t)(i - 1) * DD, xm, MM);
            // xm = seg_mean(xm, nmap, N)[nmap]
            (void)hipMemsetAsync(f_nsum, 0, (size_t)NN * DD * 4, stream);
            (void)hipMemsetAsync(f_ncnt, 0, (size_t)NN * 4, stream);
            scatter_add_kernel<<<gM32, b256, 0, stream>>>(xm, nmap, f_nsum, f_ncnt, MM);
            gather_mean_kernel<<<gM32, b256, 0, stream>>>(f_nsum, f_ncnt, nmap, tmp, MM);
            float* t2 = xm; xm = tmp; tmp = t2;
        }
        // conv: agg[dst] += (xm[src] + e)*enorm ; xm = xm + relu(agg @ W + b)
        (void)hipMemsetAsync(f_agg, 0, (size_t)MM * DD * 4, stream);
        message_kernel<<<dim3(((size_t)EE * 32 + 255) / 256), b256, 0, stream>>>(
            xm, src, dst, emap, eattr, edge_w, edge_b, f_degs, f_degd, f_agg);
        gemm(f_agg, DD, p_gnn[i], gnn_b + (size_t)i * DD, xm, xm, MM, DD, DD, 1);
    }

    // ---- patch pooling + structural PE ----
    (void)hipMemsetAsync(f_psum, 0, (size_t)BP * DD * 4, stream);
    (void)hipMemsetAsync(f_pcnt, 0, (size_t)BP * 4, stream);
    scatter_add_kernel<<<gM32, b256, 0, stream>>>(xm, sbatch, f_psum, f_pcnt, MM);
    mean_kernel<<<dim3(((size_t)BP * DD + 255) / 256), b256, 0, stream>>>(f_psum, f_pcnt, f_sub, BP);
    patchfeat_kernel<<<dim3(((size_t)BP * 32 + 255) / 256), b256, 0, stream>>>(ppos, f_patchfeat);
    gemm(f_patchfeat, 32, p_spe, spe_b, f_sub, f_sub, BP, 32, DD, 0);

    // ---- transformer ----
    tbuild_kernel<<<dim3(((size_t)RR * DD + 255) / 256), b256, 0, stream>>>(f_sub, cls, f_t);
    for (int l = 0; l < 4; ++l) {
        ln_kernel<<<dim3((RR + 7) / 8), b256, 0, stream>>>(
            f_t, n1g + (size_t)l * DD, n1b + (size_t)l * DD, f_hln, RR);
        gemm(f_hln, DD, p_qkvw[l], t_inb + (size_t)l * 384, nullptr, f_qkv, RR, DD, 384, 0);
        attn_kernel<<<dim3(BB * 4, SS), b256, 0, stream>>>(f_qkv, f_o);
        gemm(f_o, DD, p_ow[l], t_ob + (size_t)l * DD, f_t, f_t, RR, DD, DD, 0);
        ln_kernel<<<dim3((RR + 7) / 8), b256, 0, stream>>>(
            f_t, n2g + (size_t)l * DD, n2b + (size_t)l * DD, f_hln, RR);
        gemm(f_hln, DD, p_w1[l], t_b1 + (size_t)l * FFD, nullptr, f_ff, RR, DD, FFD, 2);
        gemm(f_ff, FFD, p_w2[l], t_b2 + (size_t)l * DD, f_t, f_t, RR, FFD, DD, 0);
    }

    // ---- MLP head on CLS rows (rows 0..15 of f_t) ----
    gemm(f_t, DD, p_d1, d1b, nullptr, f_h1, 16, DD, HIDD, 0);
    bnhead_kernel<<<dim3(1), b256, 0, stream>>>(f_h1, bn1g, bn1b, HIDD);
    gemm(f_h1, HIDD, p_d2, d2b, nullptr, f_h2, 16, HIDD, HIDD, 0);
    bnhead_kernel<<<dim3(1), b256, 0, stream>>>(f_h2, bn2g, bn2b, HIDD);
    gemm(f_h2, HIDD, p_d3, d3b, nullptr, (float*)d_out, 16, HIDD, OUTD, 0);
}
